// SDEBlock3D_28939489640452
// MI455X (gfx1250) — compile-verified, matrix operand order checked by reading, not run on hardware
//
#include <hip/hip_runtime.h>

#define S 64
#define S3 (S * S * S)          // 262144 voxels
#define DIMC 16                 // latent channels
#define HID 64                  // hidden channels
#define NSTEP 20
#define DT_F 0.05f
#define COEF_F 0.022360679774997897f   // SIGMA * sqrt(DT)

typedef __attribute__((ext_vector_type(16))) __bf16 v16bf;
typedef __attribute__((ext_vector_type(8)))  float  v8f;

union Frag {
    v16bf v;
    uint4 q[2];
    unsigned short u[16];
};

// round-to-nearest-even f32 -> bf16 bit pattern
__device__ __forceinline__ unsigned short f2bf(float x) {
    unsigned int u = __float_as_uint(x);
    unsigned int r = (u + 0x7FFFu + ((u >> 16) & 1u)) >> 16;
    return (unsigned short)r;
}

// 32-bit LDS byte offset of a shared-memory object (for async-to-LDS vdst operand)
__device__ __forceinline__ unsigned lds_addr(const void* p) {
    return (unsigned)(uintptr_t)(__attribute__((address_space(3))) const char*)p;
}

// async 16B copy: LDS[la] = *(saddr_base + goff); tracked on ASYNCcnt
__device__ __forceinline__ void async_cp16(unsigned la, unsigned goff, const void* base) {
    asm volatile("global_load_async_to_lds_b128 %0, %1, %2"
                 :: "v"(la), "v"(goff), "s"(base) : "memory");
}
__device__ __forceinline__ void async_wait0() {
    asm volatile("s_wait_asynccnt 0x0" ::: "memory");
}

// K-local index for element j of a lane's 16-bf16 fragment (ISA 16-bit A layout:
// lanes 0-15 hold K 0-7 (v0-3) and K 16-23 (v4-7); lanes 16-31 hold K 8-15 / 24-31)
__device__ __forceinline__ int klocal_of(int lane, int j) {
    int base = (lane >> 4) * 8;
    return (j < 8) ? (base + j) : (16 + base + (j - 8));
}

// Pack W1 [64][16][27] and W2 [16][64][27] (f32) into bf16 A-fragments in exact
// per-lane register order, so the GEMM loop reads A with one 32B LDS load.
__global__ void prep_weights(const float* __restrict__ w1, const float* __restrict__ w2,
                             unsigned short* __restrict__ A1, unsigned short* __restrict__ A2) {
    int e = blockIdx.x * blockDim.x + threadIdx.x;
    const int N1 = 4 * 14 * 512;     // 4 M-tiles x 14 K-chunks x (32 lanes x 16)
    const int N2 = 54 * 512;         // 54 K-chunks
    if (e < N1) {
        int f = e >> 9, lane = (e >> 4) & 31, j = e & 15;
        int mt = f / 14, kc = f % 14;
        int kl = klocal_of(lane, j);
        int tap = 2 * kc + (kl >> 4);       // taps paired; tap 27 is zero padding
        int ch = kl & 15;
        int m = mt * 16 + (lane & 15);
        float v = (tap < 27) ? w1[(m * DIMC + ch) * 27 + tap] : 0.0f;
        A1[e] = f2bf(v);
    } else if (e < N1 + N2) {
        int e2 = e - N1;
        int kc = e2 >> 9, lane = (e2 >> 4) & 31, j = e2 & 15;
        int kl = klocal_of(lane, j);
        int tap = kc >> 1;
        int ch = (kc & 1) * 32 + kl;        // each tap's 64 ch split in two chunks
        int m = lane & 15;
        A2[e2] = f2bf(w2[(m * HID + ch) * 27 + tap]);
    }
}

// conv1 (16->64, 3x3x3, SAME) + bias + tanh, output bf16 voxel-major [voxel][64]
__global__ __launch_bounds__(256) void conv1_tanh(
        const float* __restrict__ state, const unsigned short* __restrict__ A1,
        const float* __restrict__ b1, unsigned short* __restrict__ hbuf) {
    __shared__ __align__(16) unsigned short patch[792 * 16];    // 25344 B; reused as D stage
    __shared__ __align__(16) unsigned short a1sh[4 * 14 * 512]; // 57344 B
    const int d0 = blockIdx.x >> 5;
    const int h0 = (blockIdx.x & 31) << 1;              // 2-row tile

    // async-stage packed A1 fragments into LDS (3584 x 16B, uniform 14 iters/thread)
    {
        const unsigned abase = lds_addr(&a1sh[0]);
        for (int i = threadIdx.x; i < 3584; i += 256)
            async_cp16(abase + i * 16, i * 16, A1);
    }
    // im2col halo patch: f32 -> bf16 with SAME zero padding  [dz3][dy4][w66][ch16]
    for (int e = threadIdx.x; e < 792 * 16; e += 256) {
        int ch = e / 792, slot = e % 792;
        int dz = slot / 264, rem = slot % 264, dy = rem / 66, w = rem % 66;
        int gd = d0 + dz - 1, gh = h0 + dy - 1, gw = w - 1;
        float v = 0.0f;
        if ((unsigned)gd < S && (unsigned)gh < S && (unsigned)gw < S)
            v = state[(size_t)ch * S3 + gd * S * S + gh * S + gw];
        patch[slot * 16 + ch] = f2bf(v);
    }
    async_wait0();
    __syncthreads();

    const int lane = threadIdx.x & 31;
    const int wave = threadIdx.x >> 5;
    const int n = lane & 15, kh = lane >> 4;
    const int vox = wave * 16 + n;
    const int hy = vox >> 6, wx = vox & 63;

    v8f acc[4];
    for (int mt = 0; mt < 4; ++mt)
        for (int r = 0; r < 8; ++r) acc[mt][r] = b1[mt * 16 + r + 8 * kh];

    const uint4* a1q = reinterpret_cast<const uint4*>(a1sh);
    for (int kc = 0; kc < 14; ++kc) {
        Frag b;
        int t0 = 2 * kc, t1 = t0 + 1;
        {
            int dz = t0 / 9, r9 = t0 % 9, dy = r9 / 3, dx = r9 % 3;
            int slot = (dz * 4 + dy + hy) * 66 + wx + dx;
            b.q[0] = *reinterpret_cast<const uint4*>(&patch[slot * 16 + 8 * kh]);
        }
        if (t1 < 27) {  // uniform branch
            int dz = t1 / 9, r9 = t1 % 9, dy = r9 / 3, dx = r9 % 3;
            int slot = (dz * 4 + dy + hy) * 66 + wx + dx;
            b.q[1] = *reinterpret_cast<const uint4*>(&patch[slot * 16 + 8 * kh]);
        } else {
            b.q[1] = make_uint4(0, 0, 0, 0);
        }
        for (int mt = 0; mt < 4; ++mt) {
            Frag a;
            const uint4* p = &a1q[((mt * 14 + kc) * 32 + lane) * 2];
            a.q[0] = p[0]; a.q[1] = p[1];
            acc[mt] = __builtin_amdgcn_wmma_f32_16x16x32_bf16(
                false, a.v, false, b.v, (short)0, acc[mt], false, false);
        }
    }

    // stage D tiles (tanh -> bf16, 2 per ds_store_b32) then coalesced 16B stores
    __syncthreads();                                    // patch reads done
    unsigned* stage32 = reinterpret_cast<unsigned*>(patch);
    for (int mt = 0; mt < 4; ++mt)
        for (int r = 0; r < 8; r += 2) {
            int ch = mt * 16 + r + 8 * kh;              // even
            unsigned lo = f2bf(tanhf(acc[mt][r]));
            unsigned hi = f2bf(tanhf(acc[mt][r + 1]));
            stage32[(vox * 64 + ch) >> 1] = lo | (hi << 16);
        }
    __syncthreads();
    const uint4* sq = reinterpret_cast<const uint4*>(patch);
    uint4* hq = reinterpret_cast<uint4*>(hbuf);
    for (int e = threadIdx.x; e < 128 * 8; e += 256) {  // 4 exact iters
        int vx = e >> 3, q = e & 7;
        int row = vx >> 6, w2x = vx & 63;
        int gvox = d0 * S * S + (h0 + row) * S + w2x;
        hq[(size_t)gvox * 8 + q] = sq[e];
    }
}

// conv2 (64->16, 3x3x3, SAME) + bias, fused Euler-Maruyama update of state (f32)
__global__ __launch_bounds__(256) void conv2_update(
        const unsigned short* __restrict__ hbuf, const unsigned short* __restrict__ A2,
        const float* __restrict__ b2, const float* __restrict__ noise,
        float* __restrict__ state) {
    __shared__ __align__(16) unsigned short patch[792 * 64];  // 101376 B [slot][ch64] bf16
    __shared__ __align__(16) unsigned short a2sh[54 * 512];   // 55296 B
    const int d0 = blockIdx.x >> 5;
    const int h0 = (blockIdx.x & 31) << 1;

    // async-stage packed A2 fragments (3456 x 16B)
    {
        const unsigned abase = lds_addr(&a2sh[0]);
        for (int i = threadIdx.x; i < 3456; i += 256)
            async_cp16(abase + i * 16, i * 16, A2);
    }
    // async-stage hidden-activation halo patch; zero-fill SAME-padding slots
    {
        const unsigned pbase = lds_addr(&patch[0]);
        uint4* pq = reinterpret_cast<uint4*>(patch);
        for (int e = threadIdx.x; e < 792 * 8; e += 256) {
            int slot = e >> 3, q = e & 7;
            int dz = slot / 264, rem = slot % 264, dy = rem / 66, w = rem % 66;
            int gd = d0 + dz - 1, gh = h0 + dy - 1, gw = w - 1;
            if ((unsigned)gd < S && (unsigned)gh < S && (unsigned)gw < S) {
                unsigned goff = (unsigned)(((gd * S * S + gh * S + gw) * 8 + q) * 16);
                async_cp16(pbase + e * 16, goff, hbuf);
            } else {
                pq[e] = make_uint4(0, 0, 0, 0);
            }
        }
    }
    async_wait0();
    __syncthreads();

    const int lane = threadIdx.x & 31;
    const int wave = threadIdx.x >> 5;
    const int n = lane & 15, kh = lane >> 4;
    const int vox = wave * 16 + n;
    const int hy = vox >> 6, wx = vox & 63;

    v8f acc0, acc1;                    // two independent accumulation chains
    for (int r = 0; r < 8; ++r) { acc0[r] = b2[r + 8 * kh]; acc1[r] = 0.0f; }

    const uint4* a2q = reinterpret_cast<const uint4*>(a2sh);
    for (int tap = 0; tap < 27; ++tap) {
        int dz = tap / 9, r9 = tap % 9, dy = r9 / 3, dx = r9 % 3;
        int slot = (dz * 4 + dy + hy) * 66 + wx + dx;
        Frag b0, b1f, a0, a1f;
        b0.q[0]  = *reinterpret_cast<const uint4*>(&patch[slot * 64 + 8 * kh]);        // ch 0-31
        b0.q[1]  = *reinterpret_cast<const uint4*>(&patch[slot * 64 + 8 * kh + 16]);
        b1f.q[0] = *reinterpret_cast<const uint4*>(&patch[slot * 64 + 8 * kh + 32]);   // ch 32-63
        b1f.q[1] = *reinterpret_cast<const uint4*>(&patch[slot * 64 + 8 * kh + 48]);
        const uint4* p0 = &a2q[((tap * 2 + 0) * 32 + lane) * 2];
        const uint4* p1 = &a2q[((tap * 2 + 1) * 32 + lane) * 2];
        a0.q[0] = p0[0];  a0.q[1] = p0[1];
        a1f.q[0] = p1[0]; a1f.q[1] = p1[1];
        acc0 = __builtin_amdgcn_wmma_f32_16x16x32_bf16(
            false, a0.v, false, b0.v, (short)0, acc0, false, false);
        acc1 = __builtin_amdgcn_wmma_f32_16x16x32_bf16(
            false, a1f.v, false, b1f.v, (short)0, acc1, false, false);
    }

    const int gvox = d0 * S * S + (h0 + hy) * S + wx;
    for (int r = 0; r < 8; ++r) {
        int ch = r + 8 * kh;
        size_t idx = (size_t)ch * S3 + gvox;
        float f = acc0[r] + acc1[r];
        state[idx] = state[idx] + DT_F * f + COEF_F * noise[idx];
    }
}

extern "C" void kernel_launch(void* const* d_in, const int* in_sizes, int n_in,
                              void* d_out, int out_size, void* d_ws, size_t ws_size,
                              hipStream_t stream) {
    (void)in_sizes; (void)n_in; (void)out_size; (void)ws_size;
    const float* x     = (const float*)d_in[0];
    const float* w1    = (const float*)d_in[2];
    const float* b1    = (const float*)d_in[3];
    const float* w2    = (const float*)d_in[4];
    const float* b2    = (const float*)d_in[5];
    const float* noise = (const float*)d_in[6];

    const size_t stateBytes = (size_t)DIMC * S3 * sizeof(float);          // 16 MB
    const size_t hBytes     = (size_t)HID * S3 * sizeof(unsigned short);  // 32 MB

    float*          state = (float*)d_ws;
    unsigned short* hbuf  = (unsigned short*)((char*)d_ws + stateBytes);
    unsigned short* A1    = (unsigned short*)((char*)d_ws + stateBytes + hBytes);
    unsigned short* A2    = A1 + 4 * 14 * 512;

    hipMemcpyAsync(state, x, stateBytes, hipMemcpyDeviceToDevice, stream);
    prep_weights<<<(4 * 14 * 512 + 54 * 512 + 255) / 256, 256, 0, stream>>>(w1, w2, A1, A2);

    for (int s = 0; s < NSTEP; ++s) {
        conv1_tanh<<<2048, 256, 0, stream>>>(state, A1, b1, hbuf);
        conv2_update<<<2048, 256, 0, stream>>>(hbuf, A2, b2,
                                               noise + (size_t)s * DIMC * S3, state);
    }
    hipMemcpyAsync(d_out, state, stateBytes, hipMemcpyDeviceToDevice, stream);
}
